// BayesLinear_Normalq_35313221108256
// MI455X (gfx1250) — matrix operand order repaired and checked by reading, not compiled, not run on hardware
//
#include <hip/hip_runtime.h>
#include <hip/hip_bf16.h>

typedef __attribute__((ext_vector_type(16))) __bf16 v16bf;
typedef __attribute__((ext_vector_type(8)))  __bf16 v8bf;
typedef __attribute__((ext_vector_type(8)))  float  v8f;

#define B_DIM   8192
#define IN_DIM  4096
#define OUT_DIM 4096
#define KT      32
#define TILE_M  128
#define TILE_N  128
#define LDS_PAD 40   // row stride in bf16 elements (32 data + 8 pad = 80B, keeps 16B alignment)

__device__ __forceinline__ unsigned lds_off(const void* p) {
    // generic LDS address: high 32 = shared aperture, low 32 = workgroup LDS offset
    return (unsigned)(unsigned long long)(uintptr_t)p;
}

// ===========================================================================
// Pass 1a: X fp32 -> bf16 (64 MiB), pure streaming conversion
// ===========================================================================
__global__ __launch_bounds__(256) void cvt_x_kernel(
    const float* __restrict__ in, __bf16* __restrict__ outb, long n)
{
    long i = ((long)blockIdx.x * blockDim.x + threadIdx.x) * 8;
    const long stride = (long)gridDim.x * blockDim.x * 8;
    for (; i < n; i += stride) {
        const float4 a = *reinterpret_cast<const float4*>(in + i);
        const float4 b = *reinterpret_cast<const float4*>(in + i + 4);
        v8bf v;
        v[0] = (__bf16)a.x; v[1] = (__bf16)a.y; v[2] = (__bf16)a.z; v[3] = (__bf16)a.w;
        v[4] = (__bf16)b.x; v[5] = (__bf16)b.y; v[6] = (__bf16)b.z; v[7] = (__bf16)b.w;
        *reinterpret_cast<v8bf*>(outb + i) = v;
    }
}

// ===========================================================================
// Pass 1b: W = mu + exp(sig)*eps -> bf16 (32 MiB), fused with the weight-KL
// partial reduction (reuses exp(sig): exp(2s) = exp(s)^2). One read of the
// 192 MB parameter set serves both the GEMM operand and the KL term.
// ===========================================================================
__global__ __launch_bounds__(256) void wprep_kl_kernel(
    const float* __restrict__ mu, const float* __restrict__ sig,
    const float* __restrict__ eps, __bf16* __restrict__ wb,
    double* __restrict__ partial, long n)
{
    __shared__ double red[256];
    float kls = 0.f;
    long i = ((long)blockIdx.x * 256 + threadIdx.x) * 8;
    const long stride = (long)gridDim.x * 256 * 8;
    for (; i < n; i += stride) {
        const float4 m0 = *reinterpret_cast<const float4*>(mu  + i);
        const float4 m1 = *reinterpret_cast<const float4*>(mu  + i + 4);
        const float4 s0 = *reinterpret_cast<const float4*>(sig + i);
        const float4 s1 = *reinterpret_cast<const float4*>(sig + i + 4);
        const float4 e0 = *reinterpret_cast<const float4*>(eps + i);
        const float4 e1 = *reinterpret_cast<const float4*>(eps + i + 4);
        const float mv[8] = {m0.x,m0.y,m0.z,m0.w,m1.x,m1.y,m1.z,m1.w};
        const float sv[8] = {s0.x,s0.y,s0.z,s0.w,s1.x,s1.y,s1.z,s1.w};
        const float ev[8] = {e0.x,e0.y,e0.z,e0.w,e1.x,e1.y,e1.z,e1.w};
        v8bf w;
#pragma unroll
        for (int c = 0; c < 8; ++c) {
            const float ex = __expf(sv[c]);
            w[c] = (__bf16)(mv[c] + ex * ev[c]);
            kls += 1.0f + 2.0f * sv[c] - mv[c] * mv[c] - ex * ex;
        }
        *reinterpret_cast<v8bf*>(wb + i) = w;
    }
    red[threadIdx.x] = (double)kls;
    __syncthreads();
    for (int off = 128; off > 0; off >>= 1) {
        if (threadIdx.x < off) red[threadIdx.x] += red[threadIdx.x + off];
        __syncthreads();
    }
    if (threadIdx.x == 0) partial[blockIdx.x] = red[0];
}

// ===========================================================================
// Pass 2: GEMM on pre-converted bf16 operands. LDS staging via
// GLOBAL_LOAD_ASYNC_TO_LDS_B128 (ASYNCcnt), double-buffered; the
// s_wait_asynccnt sits *behind* the 8 WMMAs of the current tile.
// ===========================================================================
__global__ __launch_bounds__(256) void bayes_gemm_pre(
    const __bf16* __restrict__ xb, const __bf16* __restrict__ wb,
    const float* __restrict__ bmu, const float* __restrict__ bsig,
    const float* __restrict__ epsb, float* __restrict__ out)
{
    __shared__ __align__(16) __bf16 Xs[2][TILE_M][LDS_PAD];
    __shared__ __align__(16) __bf16 Ws[2][TILE_N][LDS_PAD];

    const int t     = threadIdx.x;
    const int wave  = t >> 5;
    const int lane  = t & 31;
    const int h     = lane >> 4;
    const int ln    = lane & 15;
    const int waveM = wave >> 1;   // 0..3
    const int waveN = wave & 1;    // 0..1
    const int mBase = blockIdx.x * TILE_M;
    const int nBase = blockIdx.y * TILE_N;

    // staging map: thread t owns 16 contiguous bf16 (32B) of one tile row
    const int srow  = t >> 1;          // 0..127
    const int skoff = (t & 1) * 16;    // 0 or 16 elements

    v8f acc[2][4];
#pragma unroll
    for (int i = 0; i < 2; ++i)
#pragma unroll
        for (int j = 0; j < 4; ++j)
            acc[i][j] = v8f{0.f,0.f,0.f,0.f,0.f,0.f,0.f,0.f};

    auto stage = [&](int kt, int buf) {
        const long kb = (long)kt * KT + skoff;
        const __bf16* xg = xb + (long)(mBase + srow) * IN_DIM + kb;
        const __bf16* wg = wb + (long)(nBase + srow) * IN_DIM + kb;
        const unsigned xl = lds_off(&Xs[buf][srow][skoff]);
        const unsigned wl = lds_off(&Ws[buf][srow][skoff]);
        // IOFFSET applies to both LDS and global sides (ISA 15.18.3), and both
        // sides are contiguous here, so two 16B chunks per 32B span.
        asm volatile("global_load_async_to_lds_b128 %0, %1, off"
                     :: "v"(xl), "v"(xg) : "memory");
        asm volatile("global_load_async_to_lds_b128 %0, %1, off offset:16"
                     :: "v"(xl), "v"(xg) : "memory");
        asm volatile("global_load_async_to_lds_b128 %0, %1, off"
                     :: "v"(wl), "v"(wg) : "memory");
        asm volatile("global_load_async_to_lds_b128 %0, %1, off offset:16"
                     :: "v"(wl), "v"(wg) : "memory");
    };

    auto compute = [&](int buf) {
        v16bf afrag[2], bfrag[4];
#pragma unroll
        for (int mt = 0; mt < 2; ++mt) {
            // A 16x32 bf16: lanes 0-15 hold K{0..7,16..23}, lanes 16-31 K{8..15,24..31}
            const __bf16* ap = &Xs[buf][waveM * 32 + mt * 16 + ln][h * 8];
            union { v16bf v; v8bf p[2]; } ua;
            ua.p[0] = *reinterpret_cast<const v8bf*>(ap);
            ua.p[1] = *reinterpret_cast<const v8bf*>(ap + 16);
            afrag[mt] = ua.v;
        }
#pragma unroll
        for (int nt = 0; nt < 4; ++nt) {
            // B 32x16 bf16: lanes 0-15 hold K{0..7,16..23} (N=lane), lanes 16-31 K{8..15,24..31}
            const __bf16* bp = &Ws[buf][waveN * 64 + nt * 16 + ln][h * 8];
            union { v16bf v; v8bf p[2]; } ub;
            ub.p[0] = *reinterpret_cast<const v8bf*>(bp);
            ub.p[1] = *reinterpret_cast<const v8bf*>(bp + 16);
            bfrag[nt] = ub.v;
        }
#pragma unroll
        for (int mt = 0; mt < 2; ++mt)
#pragma unroll
            for (int nt = 0; nt < 4; ++nt)
                acc[mt][nt] = __builtin_amdgcn_wmma_f32_16x16x32_bf16(
                    false, afrag[mt], false, bfrag[nt],
                    (short)0, acc[mt][nt], false, false);
    };

    const int NKT = IN_DIM / KT;  // 128

    stage(0, 0);
    asm volatile("s_wait_asynccnt 0x0" ::: "memory");
    __syncthreads();

    for (int kt = 0; kt < NKT; ++kt) {
        const int buf = kt & 1;
        if (kt + 1 < NKT) stage(kt + 1, buf ^ 1);   // async fill of other buffer
        compute(buf);                                // 8x wmma over current buffer
        if (kt + 1 < NKT)
            asm volatile("s_wait_asynccnt 0x0" ::: "memory");
        __syncthreads();
    }

#pragma unroll
    for (int nt = 0; nt < 4; ++nt) {
        const int c = nBase + waveN * 64 + nt * 16 + ln;
        const float bias = bmu[c] + __expf(bsig[c]) * epsb[c];
#pragma unroll
        for (int mt = 0; mt < 2; ++mt) {
            const int rbase = mBase + waveM * 32 + mt * 16 + h * 8;
#pragma unroll
            for (int g = 0; g < 8; ++g)
                out[(long)(rbase + g) * OUT_DIM + c] = acc[mt][nt][g] + bias;
        }
    }
}

// ===========================================================================
// Fallback fused GEMM (round-1 version) for small-ws harnesses
// ===========================================================================
__global__ __launch_bounds__(256) void bayes_gemm_fused(
    const float* __restrict__ x,
    const float* __restrict__ wmu,
    const float* __restrict__ wsig,
    const float* __restrict__ bmu,
    const float* __restrict__ bsig,
    const float* __restrict__ epsw,
    const float* __restrict__ epsb,
    float* __restrict__ out)
{
    __shared__ __align__(16) __bf16 Xs[2][TILE_M][LDS_PAD];
    __shared__ __align__(16) __bf16 Ws[2][TILE_N][LDS_PAD];

    const int t     = threadIdx.x;
    const int wave  = t >> 5;
    const int lane  = t & 31;
    const int h     = lane >> 4;
    const int ln    = lane & 15;
    const int waveM = wave >> 1;
    const int waveN = wave & 1;
    const int mBase = blockIdx.x * TILE_M;
    const int nBase = blockIdx.y * TILE_N;
    const int srow  = t >> 1;
    const int skoff = (t & 1) * 16;

    v8f acc[2][4];
#pragma unroll
    for (int i = 0; i < 2; ++i)
#pragma unroll
        for (int j = 0; j < 4; ++j)
            acc[i][j] = v8f{0.f,0.f,0.f,0.f,0.f,0.f,0.f,0.f};

    float4 xr[4], mr[4], sr[4], er[4];

    auto ldglobal = [&](int kt) {
        const long kb = (long)kt * KT + skoff;
        const float4* xp = reinterpret_cast<const float4*>(x    + (long)(mBase + srow) * IN_DIM + kb);
        const long wrow  = (long)(nBase + srow) * IN_DIM + kb;
        const float4* mp = reinterpret_cast<const float4*>(wmu  + wrow);
        const float4* sp = reinterpret_cast<const float4*>(wsig + wrow);
        const float4* ep = reinterpret_cast<const float4*>(epsw + wrow);
#pragma unroll
        for (int j = 0; j < 4; ++j) { xr[j] = xp[j]; mr[j] = mp[j]; sr[j] = sp[j]; er[j] = ep[j]; }
    };

    auto stlds = [&](int buf) {
        v8bf vx0, vx1, vw0, vw1;
#pragma unroll
        for (int j = 0; j < 4; ++j) {
            const float xv[4] = {xr[j].x, xr[j].y, xr[j].z, xr[j].w};
            const float mv[4] = {mr[j].x, mr[j].y, mr[j].z, mr[j].w};
            const float sv[4] = {sr[j].x, sr[j].y, sr[j].z, sr[j].w};
            const float ev[4] = {er[j].x, er[j].y, er[j].z, er[j].w};
#pragma unroll
            for (int c = 0; c < 4; ++c) {
                const int e = j * 4 + c;
                const __bf16 xb = (__bf16)xv[c];
                const __bf16 wb = (__bf16)(mv[c] + __expf(sv[c]) * ev[c]);
                if (e < 8) { vx0[e] = xb;     vw0[e] = wb; }
                else       { vx1[e - 8] = xb; vw1[e - 8] = wb; }
            }
        }
        *reinterpret_cast<v8bf*>(&Xs[buf][srow][skoff])     = vx0;
        *reinterpret_cast<v8bf*>(&Xs[buf][srow][skoff + 8]) = vx1;
        *reinterpret_cast<v8bf*>(&Ws[buf][srow][skoff])     = vw0;
        *reinterpret_cast<v8bf*>(&Ws[buf][srow][skoff + 8]) = vw1;
    };

    auto compute = [&](int buf) {
        v16bf afrag[2], bfrag[4];
#pragma unroll
        for (int mt = 0; mt < 2; ++mt) {
            const __bf16* ap = &Xs[buf][waveM * 32 + mt * 16 + ln][h * 8];
            union { v16bf v; v8bf p[2]; } ua;
            ua.p[0] = *reinterpret_cast<const v8bf*>(ap);
            ua.p[1] = *reinterpret_cast<const v8bf*>(ap + 16);
            afrag[mt] = ua.v;
        }
#pragma unroll
        for (int nt = 0; nt < 4; ++nt) {
            const __bf16* bp = &Ws[buf][waveN * 64 + nt * 16 + ln][h * 8];
            union { v16bf v; v8bf p[2]; } ub;
            ub.p[0] = *reinterpret_cast<const v8bf*>(bp);
            ub.p[1] = *reinterpret_cast<const v8bf*>(bp + 16);
            bfrag[nt] = ub.v;
        }
#pragma unroll
        for (int mt = 0; mt < 2; ++mt)
#pragma unroll
            for (int nt = 0; nt < 4; ++nt)
                acc[mt][nt] = __builtin_amdgcn_wmma_f32_16x16x32_bf16(
                    false, afrag[mt], false, bfrag[nt],
                    (short)0, acc[mt][nt], false, false);
    };

    const int NKT = IN_DIM / KT;

    ldglobal(0);
    stlds(0);
    __syncthreads();

    for (int kt = 0; kt < NKT; ++kt) {
        const int buf = kt & 1;
        if (kt + 1 < NKT) ldglobal(kt + 1);
        compute(buf);
        if (kt + 1 < NKT) stlds(buf ^ 1);
        __syncthreads();
    }

#pragma unroll
    for (int nt = 0; nt < 4; ++nt) {
        const int c = nBase + waveN * 64 + nt * 16 + ln;
        const float bias = bmu[c] + __expf(bsig[c]) * epsb[c];
#pragma unroll
        for (int mt = 0; mt < 2; ++mt) {
            const int rbase = mBase + waveM * 32 + mt * 16 + h * 8;
#pragma unroll
            for (int g = 0; g < 8; ++g)
                out[(long)(rbase + g) * OUT_DIM + c] = acc[mt][nt][g] + bias;
        }
    }
}

// ===========================================================================
// KL reductions
// ===========================================================================
__global__ __launch_bounds__(256) void kl_partial_kernel(
    const float* __restrict__ mu, const float* __restrict__ sig,
    double* __restrict__ partial, long n)
{
    __shared__ double red[256];
    long i = (long)blockIdx.x * blockDim.x + threadIdx.x;
    const long stride = (long)gridDim.x * blockDim.x;
    float s = 0.f;
    for (; i < n; i += stride) {
        const float m = mu[i], g = sig[i];
        s += 1.0f + 2.0f * g - m * m - __expf(2.0f * g);
    }
    red[threadIdx.x] = (double)s;
    __syncthreads();
    for (int off = 128; off > 0; off >>= 1) {
        if (threadIdx.x < off) red[threadIdx.x] += red[threadIdx.x + off];
        __syncthreads();
    }
    if (threadIdx.x == 0) partial[blockIdx.x] = red[0];
}

__global__ __launch_bounds__(256) void kl_final_kernel(
    const double* __restrict__ partial, int np,
    const float* __restrict__ bmu, const float* __restrict__ bsig,
    const float* __restrict__ epsb_unused, int nb,
    float* __restrict__ out_scalar)
{
    (void)epsb_unused;
    __shared__ double red[256];
    double s = 0.0;
    for (int i = threadIdx.x; i < np; i += 256) s += partial[i];
    float sb = 0.f;
    for (int i = threadIdx.x; i < nb; i += 256) {
        const float m = bmu[i], g = bsig[i];
        sb += 1.0f + 2.0f * g - m * m - __expf(2.0f * g);
    }
    red[threadIdx.x] = s + (double)sb;
    __syncthreads();
    for (int off = 128; off > 0; off >>= 1) {
        if (threadIdx.x < off) red[threadIdx.x] += red[threadIdx.x + off];
        __syncthreads();
    }
    if (threadIdx.x == 0) *out_scalar = (float)(-0.5 * red[0]);
}

// ===========================================================================
extern "C" void kernel_launch(void* const* d_in, const int* in_sizes, int n_in,
                              void* d_out, int out_size, void* d_ws, size_t ws_size,
                              hipStream_t stream)
{
    (void)in_sizes; (void)n_in; (void)out_size;
    const float* x    = (const float*)d_in[0];
    const float* wmu  = (const float*)d_in[1];
    const float* wsig = (const float*)d_in[2];
    const float* bmu  = (const float*)d_in[3];
    const float* bsig = (const float*)d_in[4];
    const float* epsw = (const float*)d_in[5];
    const float* epsb = (const float*)d_in[6];
    float* out = (float*)d_out;

    const long nW = (long)OUT_DIM * IN_DIM;   // 16.7M
    const long nX = (long)B_DIM * IN_DIM;     // 33.5M

    // ws layout: [0,8KB) partials, [64KB, 64KB+64MiB) Xbf16, then 32MiB Wbf16
    double* partial = (double*)d_ws;
    const size_t xb_off = 65536;
    const size_t wb_off = xb_off + (size_t)nX * sizeof(__bf16);
    const size_t need   = wb_off + (size_t)nW * sizeof(__bf16);

    dim3 grid(B_DIM / TILE_M, OUT_DIM / TILE_N);   // 64 x 32, M fastest for L2 reuse of W

    if (ws_size >= need) {
        __bf16* xbuf = (__bf16*)((char*)d_ws + xb_off);
        __bf16* wbuf = (__bf16*)((char*)d_ws + wb_off);
        cvt_x_kernel<<<4096, 256, 0, stream>>>(x, xbuf, nX);
        wprep_kl_kernel<<<1024, 256, 0, stream>>>(wmu, wsig, epsw, wbuf, partial, nW);
        bayes_gemm_pre<<<grid, 256, 0, stream>>>(xbuf, wbuf, bmu, bsig, epsb, out);
        kl_final_kernel<<<1, 256, 0, stream>>>(partial, 1024, bmu, bsig, nullptr, OUT_DIM,
                                               out + (long)B_DIM * OUT_DIM);
    } else {
        bayes_gemm_fused<<<grid, 256, 0, stream>>>(x, wmu, wsig, bmu, bsig, epsw, epsb, out);
        kl_partial_kernel<<<1024, 256, 0, stream>>>(wmu, wsig, partial, nW);
        kl_final_kernel<<<1, 256, 0, stream>>>(partial, 1024, bmu, bsig, nullptr, OUT_DIM,
                                               out + (long)B_DIM * OUT_DIM);
    }
}
